// MultiHeadAttention_69535520522525
// MI455X (gfx1250) — matrix operand order repaired and checked by reading, not compiled, not run on hardware
//
#include <hip/hip_runtime.h>
#include <hip/hip_bf16.h>

typedef __attribute__((ext_vector_type(16))) __bf16 v16bf;
typedef __attribute__((ext_vector_type(8)))  __bf16 v8bf;
typedef __attribute__((ext_vector_type(8)))  float  v8f;

#define WMMA_BF16(a, b, c) \
  __builtin_amdgcn_wmma_f32_16x16x32_bf16(false, (a), false, (b), (short)0, (c), false, false)

union ABFrag { v16bf v; v8bf h[2]; };

static constexpr int Bc = 4, Sc = 2048, Dc = 1024;
static constexpr int Mc = Bc * Sc;            // 8192 tokens

// ---------------------------------------------------------------------------
// f32 -> bf16 conversion (grid-stride)
// ---------------------------------------------------------------------------
__global__ void cvt_bf16(const float* __restrict__ in, __bf16* __restrict__ out, int n) {
  int i = blockIdx.x * blockDim.x + threadIdx.x;
  int stride = gridDim.x * blockDim.x;
  for (; i < n; i += stride) out[i] = (__bf16)in[i];
}

// ---------------------------------------------------------------------------
// Weight transpose + downconvert: WT[k][n] = (bf16) W[n][k], both 1024x1024
// ---------------------------------------------------------------------------
__global__ __launch_bounds__(256) void wtrans(const float* __restrict__ W,
                                              __bf16* __restrict__ WT) {
  __shared__ float tile[32][33];
  int k0 = blockIdx.x * 32, n0 = blockIdx.y * 32;
  int tx = threadIdx.x, ty = threadIdx.y;          // 32 x 8
  for (int r = 0; r < 32; r += 8)
    tile[ty + r][tx] = W[(size_t)(n0 + ty + r) * Dc + (k0 + tx)];
  __syncthreads();
  for (int r = 0; r < 32; r += 8)
    WT[(size_t)(k0 + ty + r) * Dc + (n0 + tx)] = (__bf16)tile[tx][ty + r];
}

// ---------------------------------------------------------------------------
// GEMM: C[m,n] = sum_k A[m,k] * WT[k,n] + bias[n]
//   A:  bf16 [8192, 1024] row-major
//   WT: bf16 [1024, 1024] row-major (pre-transposed weight)
//   mode 0: f32 out, row-major [8192,1024]   (final projection -> d_out)
//   mode 1: bf16 out, [B,H,S,Hd]             (q / v projections)
//   mode 2: bf16 out, [B,H,Hd,S]             (k projection, transposed)
// Block = 128 threads (4 waves, 2x2); each wave computes a 32x32 tile via
// 2x2 register blocking (4 WMMAs per K-step, 1:1 WMMA:b128-load ratio; B loaded per element).
// ---------------------------------------------------------------------------
__global__ __launch_bounds__(128) void gemm_bf16(const __bf16* __restrict__ A,
                                                 const __bf16* __restrict__ WT,
                                                 const float* __restrict__ bias,
                                                 void* __restrict__ outp, int mode) {
  const int K = 1024, N = 1024;
  int lane = threadIdx.x & 31;
  int wave = threadIdx.x >> 5;     // 0..3
  int wx = wave & 1;               // N direction
  int wy = wave >> 1;              // M direction
  int half = lane >> 4;
  int l15  = lane & 15;
  int n0 = blockIdx.x * 64 + wx * 32;   // wave covers n0 .. n0+31
  int m0 = blockIdx.y * 64 + wy * 32;   // wave covers m0 .. m0+31

  const __bf16* arow0 = A + (size_t)(m0 + l15) * K;
  const __bf16* arow1 = A + (size_t)(m0 + 16 + l15) * K;

  v8f acc[2][2];
  for (int i = 0; i < 2; i++)
    for (int j = 0; j < 2; j++)
      acc[i][j] = (v8f){0.f, 0.f, 0.f, 0.f, 0.f, 0.f, 0.f, 0.f};

  for (int kb = 0; kb < K; kb += 32) {
    ABFrag a0, a1;
    a0.h[0] = *(const v8bf*)(arow0 + kb + 8 * half);
    a0.h[1] = *(const v8bf*)(arow0 + kb + 16 + 8 * half);
    a1.h[0] = *(const v8bf*)(arow1 + kb + 8 * half);
    a1.h[1] = *(const v8bf*)(arow1 + kb + 16 + 8 * half);
    v16bf b0, b1;
    for (int s = 0; s < 16; s++) {
      int kk = kb + ((s < 8) ? (8 * half + s) : (16 + 8 * half + s - 8));
      b0[s] = WT[(size_t)kk * N + n0 + l15];
      b1[s] = WT[(size_t)kk * N + n0 + 16 + l15];
    }
    acc[0][0] = WMMA_BF16(a0.v, b0, acc[0][0]);
    acc[0][1] = WMMA_BF16(a0.v, b1, acc[0][1]);
    acc[1][0] = WMMA_BF16(a1.v, b0, acc[1][0]);
    acc[1][1] = WMMA_BF16(a1.v, b1, acc[1][1]);
  }

  for (int ni = 0; ni < 2; ni++) {
    int col = n0 + 16 * ni + l15;
    float bv = bias[col];
    for (int mi = 0; mi < 2; mi++) {
      if (mode == 0) {
        float* out = (float*)outp;
        for (int j = 0; j < 8; j++) {
          int row = m0 + 16 * mi + j + 8 * half;
          out[(size_t)row * N + col] = acc[mi][ni][j] + bv;
        }
      } else {
        __bf16* out = (__bf16*)outp;
        int h = col >> 6, hd = col & 63;
        for (int j = 0; j < 8; j++) {
          int row = m0 + 16 * mi + j + 8 * half;
          int b = row >> 11, s = row & 2047;
          size_t idx = (mode == 1)
              ? ((((size_t)b * 16 + h) * 2048 + s) * 64 + hd)
              : ((((size_t)b * 16 + h) * 64 + hd) * 2048 + s);
          out[idx] = (__bf16)(acc[mi][ni][j] + bv);
        }
      }
    }
  }
}

// ---------------------------------------------------------------------------
// Flash attention per (b,h): qp [BH,S,64], kp [BH,64,S] (transposed), vp [BH,S,64]
// Output: ao bf16 [B,S,1024] (heads concatenated).
// Block = 4 waves; each wave owns 16 queries, streams keys 32 at a time.
// ---------------------------------------------------------------------------
__global__ __launch_bounds__(128) void flash_attn(const __bf16* __restrict__ qp,
                                                  const __bf16* __restrict__ kp,
                                                  const __bf16* __restrict__ vp,
                                                  __bf16* __restrict__ ao) {
  __shared__ __align__(16) __bf16 Pst[4][16][32];
  int lane = threadIdx.x & 31;
  int wave = threadIdx.x >> 5;
  int half = lane >> 4;
  int l15  = lane & 15;
  int bh = blockIdx.x;
  int q0 = blockIdx.y * 64 + wave * 16;

  const __bf16* qb = qp + (size_t)bh * Sc * 64;
  const __bf16* kb = kp + (size_t)bh * 64 * Sc;
  const __bf16* vb = vp + (size_t)bh * Sc * 64;

  // q fragments (A-matrix layout), Hd=64 -> two K=32 fragments
  ABFrag qf[2];
  {
    const __bf16* qrow = qb + (size_t)(q0 + l15) * 64;
    for (int d = 0; d < 2; d++) {
      qf[d].h[0] = *(const v8bf*)(qrow + 32 * d + 8 * half);
      qf[d].h[1] = *(const v8bf*)(qrow + 32 * d + 16 + 8 * half);
    }
  }

  float mrun[8], lrun[8];
  v8f oacc[4];
  for (int j = 0; j < 8; j++) { mrun[j] = -1e30f; lrun[j] = 0.f; }
  for (int t = 0; t < 4; t++) oacc[t] = (v8f){0.f, 0.f, 0.f, 0.f, 0.f, 0.f, 0.f, 0.f};

  const float scale = 0.125f;  // 1/sqrt(64)

  for (int kb0 = 0; kb0 < Sc; kb0 += 32) {
    // two 16-key score tiles, online softmax, stage P into LDS (A layout source)
    for (int t = 0; t < 2; t++) {
      int sk = kb0 + 16 * t;
      v8f sacc = {0.f, 0.f, 0.f, 0.f, 0.f, 0.f, 0.f, 0.f};
      for (int d = 0; d < 2; d++) {
        // B fragment: lane = key (l15), elems = d index within K=32 slab (table order)
        v16bf kf;
        for (int s = 0; s < 16; s++) {
          int dd = 32 * d + ((s < 8) ? (8 * half + s) : (16 + 8 * half + s - 8));
          kf[s] = kb[(size_t)dd * Sc + sk + l15];
        }
        sacc = WMMA_BF16(qf[d].v, kf, sacc);
      }
      for (int j = 0; j < 8; j++) {
        float sv = sacc[j] * scale;
        float tm = sv;
        for (int off = 1; off < 16; off <<= 1)
          tm = fmaxf(tm, __shfl_xor(tm, off, 32));
        float mnew = fmaxf(mrun[j], tm);
        float p    = __expf(sv - mnew);
        float corr = __expf(mrun[j] - mnew);
        float ts = p;
        for (int off = 1; off < 16; off <<= 1)
          ts += __shfl_xor(ts, off, 32);
        lrun[j] = lrun[j] * corr + ts;
        mrun[j] = mnew;
        for (int t4 = 0; t4 < 4; t4++) oacc[t4][j] *= corr;
        Pst[wave][j + 8 * half][16 * t + l15] = (__bf16)p;
      }
    }
    asm volatile("s_wait_dscnt 0x0" ::: "memory");
    // Reload P as A-matrix fragment (16 queries x 32 keys)
    ABFrag pf;
    pf.h[0] = *(const v8bf*)&Pst[wave][l15][8 * half];
    pf.h[1] = *(const v8bf*)&Pst[wave][l15][16 + 8 * half];
    for (int t4 = 0; t4 < 4; t4++) {
      v16bf vf;
      for (int s = 0; s < 16; s++) {
        int kk = kb0 + ((s < 8) ? (8 * half + s) : (16 + 8 * half + s - 8));
        vf[s] = vb[(size_t)kk * 64 + 16 * t4 + l15];
      }
      oacc[t4] = WMMA_BF16(pf.v, vf, oacc[t4]);
    }
  }

  int b = bh >> 4, h = bh & 15;
  for (int t4 = 0; t4 < 4; t4++) {
    int col = h * 64 + 16 * t4 + l15;
    for (int j = 0; j < 8; j++) {
      int s = q0 + j + 8 * half;
      float v = oacc[t4][j] / lrun[j];
      ao[((size_t)b * Sc + s) * 1024 + col] = (__bf16)v;
    }
  }
}

// ---------------------------------------------------------------------------
extern "C" void kernel_launch(void* const* d_in, const int* in_sizes, int n_in,
                              void* d_out, int out_size, void* d_ws, size_t ws_size,
                              hipStream_t stream) {
  (void)in_sizes; (void)n_in; (void)out_size; (void)ws_size;
  const float* Q  = (const float*)d_in[0];
  const float* Kx = (const float*)d_in[1];
  const float* V  = (const float*)d_in[2];
  const float* Wq = (const float*)d_in[3];
  const float* bq = (const float*)d_in[4];
  const float* Wk = (const float*)d_in[5];
  const float* bk = (const float*)d_in[6];
  const float* Wv = (const float*)d_in[7];
  const float* bv = (const float*)d_in[8];
  const float* Wo = (const float*)d_in[9];
  const float* bo = (const float*)d_in[10];
  float* out = (float*)d_out;

  char* ws = (char*)d_ws;
  const size_t MB = 1ull << 20;
  __bf16* WTq = (__bf16*)(ws + 0 * MB);
  __bf16* WTk = (__bf16*)(ws + 2 * MB);
  __bf16* WTv = (__bf16*)(ws + 4 * MB);
  __bf16* WTo = (__bf16*)(ws + 6 * MB);
  __bf16* Qb  = (__bf16*)(ws + 8 * MB);
  __bf16* Kb  = (__bf16*)(ws + 24 * MB);
  __bf16* Vb  = (__bf16*)(ws + 40 * MB);
  __bf16* qpj = (__bf16*)(ws + 56 * MB);   // [B,H,S,Hd]
  __bf16* kpj = (__bf16*)(ws + 72 * MB);   // [B,H,Hd,S]
  __bf16* vpj = (__bf16*)(ws + 88 * MB);   // [B,H,S,Hd]
  __bf16* aob = (__bf16*)(ws + 104 * MB);  // [B,S,D]

  dim3 tb(32, 8), tg(32, 32);
  wtrans<<<tg, tb, 0, stream>>>(Wq, WTq);
  wtrans<<<tg, tb, 0, stream>>>(Wk, WTk);
  wtrans<<<tg, tb, 0, stream>>>(Wv, WTv);
  wtrans<<<tg, tb, 0, stream>>>(Wo, WTo);

  int n = Mc * Dc;  // 8388608
  cvt_bf16<<<2048, 256, 0, stream>>>(Q,  Qb, n);
  cvt_bf16<<<2048, 256, 0, stream>>>(Kx, Kb, n);
  cvt_bf16<<<2048, 256, 0, stream>>>(V,  Vb, n);

  dim3 gg(16, 128);  // N/64 x M/64
  gemm_bf16<<<gg, 128, 0, stream>>>(Qb, WTq, bq, qpj, 1);
  gemm_bf16<<<gg, 128, 0, stream>>>(Kb, WTk, bk, kpj, 2);
  gemm_bf16<<<gg, 128, 0, stream>>>(Vb, WTv, bv, vpj, 1);

  flash_attn<<<dim3(64, 32), 128, 0, stream>>>(qpj, kpj, vpj, aob);

  gemm_bf16<<<gg, 128, 0, stream>>>(aob, WTo, bo, (void*)out, 0);
}
